// VanillaRNN_17772574671228
// MI455X (gfx1250) — matrix operand order repaired and checked by reading, not compiled, not run on hardware
//
#include <hip/hip_runtime.h>
#include <hip/hip_bf16.h>

// ---- CDNA5 WMMA vector types ----
typedef __attribute__((ext_vector_type(16))) __bf16 v16bf;
typedef __attribute__((ext_vector_type(8)))  __bf16 v8bf;
typedef __attribute__((ext_vector_type(8)))  float  v8f;
typedef __attribute__((ext_vector_type(4)))  float  v4f;

#define T_SEQ   2048
#define BATCH   512
#define HID     256
#define CLS     10
#define NB      16          // batch columns per workgroup (== WMMA N)
#define KSTEPS  (HID / 32)  // 8 K-slices of 32 for 16x16x32 bf16 WMMA
#define NWAVE   8
#define MT_PER_WAVE ((HID / 16) / NWAVE)  // 2 M-tiles per wave

__device__ __forceinline__ float fast_tanh(float v) {
    // tanh(v) = 1 - 2 / (exp(2v) + 1); exp via v_exp_f32, div via v_rcp_f32.
    float e = __builtin_amdgcn_exp2f(v * 2.8853900817779268f);  // 2*log2(e)
    return 1.0f - 2.0f * __builtin_amdgcn_rcpf(e + 1.0f);
}

__global__ __launch_bounds__(256) void VanillaRNN_cdna5_kernel(
    const float* __restrict__ x,    // [B, T]
    const float* __restrict__ Whx,  // [H]
    const float* __restrict__ Whh,  // [H, H] row-major (m, k)
    const float* __restrict__ Wph,  // [C, H]
    const float* __restrict__ bh,   // [H]
    const float* __restrict__ bp,   // [C]
    float* __restrict__ out)        // [B, C]
{
    // Double-buffered hidden state, bf16, B-operand friendly: [buf][b][h]
    __shared__ __align__(16) __bf16 hbuf[2][NB][HID];  // 16 KB

    const int tid   = threadIdx.x;
    const int wave  = tid >> 5;
    const int lane  = tid & 31;
    const int sel   = lane >> 4;    // half-wave select
    const int l15   = lane & 15;
    const int bcol  = l15;          // batch column within this WG's tile
    const int bbase = blockIdx.x * NB;

    // ---- Preload W_hh as bf16 A-fragments into registers (time-invariant) ----
    // A 16x32 layout (contiguous K, matches B): lanes 0-15 hold K {kb..kb+15},
    // lanes 16-31 hold K {kb+16..kb+31} for row M = l15.
    v16bf afrag[MT_PER_WAVE][KSTEPS];
    #pragma unroll
    for (int tt = 0; tt < MT_PER_WAVE; ++tt) {
        const int m = (wave * MT_PER_WAVE + tt) * 16 + l15;
        #pragma unroll
        for (int ks = 0; ks < KSTEPS; ++ks) {
            const int kb = ks * 32 + sel * 16;
            v4f lo0 = *(const v4f*)&Whh[m * HID + kb];
            v4f lo1 = *(const v4f*)&Whh[m * HID + kb + 4];
            v4f hi0 = *(const v4f*)&Whh[m * HID + kb + 8];
            v4f hi1 = *(const v4f*)&Whh[m * HID + kb + 12];
            v16bf f;
            #pragma unroll
            for (int i = 0; i < 4; ++i) {
                f[i]      = (__bf16)lo0[i];
                f[4 + i]  = (__bf16)lo1[i];
                f[8 + i]  = (__bf16)hi0[i];
                f[12 + i] = (__bf16)hi1[i];
            }
            afrag[tt][ks] = f;
        }
    }

    // ---- Per-tile W_hx / bias_h slices: rows M = tilebase + 8*sel + (0..7) ----
    v4f whxA[MT_PER_WAVE], whxB[MT_PER_WAVE], bhA[MT_PER_WAVE], bhB[MT_PER_WAVE];
    #pragma unroll
    for (int tt = 0; tt < MT_PER_WAVE; ++tt) {
        const int mb = (wave * MT_PER_WAVE + tt) * 16 + sel * 8;
        whxA[tt] = *(const v4f*)&Whx[mb];
        whxB[tt] = *(const v4f*)&Whx[mb + 4];
        bhA[tt]  = *(const v4f*)&bh[mb];
        bhB[tt]  = *(const v4f*)&bh[mb + 4];
    }

    // ---- h0 = 0 ----
    for (int i = tid; i < NB * HID; i += 256)
        ((__bf16*)hbuf[0])[i] = (__bf16)0.0f;
    __syncthreads();

    const float* xrow = &x[(bbase + bcol) * T_SEQ];

    // Software-pipeline x: the global load for step t+1 is issued at the top
    // of step t, so its L2/HBM latency hides under 16 WMMA + 18 DS ops +
    // the tanh chain + the barrier instead of stalling the epilogue.
    float xt = xrow[0];

    // ---- sequential time loop ----
    for (int t = 0; t < T_SEQ; ++t) {
        const int cur = t & 1, nxt = cur ^ 1;
        if ((t & 63) == 0)  // pull the next x cacheline early: global_prefetch_b8
            __builtin_prefetch(&xrow[t + 64], 0, 0);
        const int tn = (t + 1 < T_SEQ) ? (t + 1) : (T_SEQ - 1);
        const float xnext = xrow[tn];   // issued now, consumed next iteration

        v8f acc0 = {};
        v8f acc1 = {};
        #pragma unroll
        for (int ks = 0; ks < KSTEPS; ++ks) {
            // B-operand (h_old): lane n -> column N=n, contiguous K sel*16..sel*16+15
            const int kb = ks * 32 + sel * 16;
            v16bf bf = *(const v16bf*)&hbuf[cur][bcol][kb];  // 2x ds_load_b128
            acc0 = __builtin_amdgcn_wmma_f32_16x16x32_bf16(
                       false, afrag[0][ks], false, bf, (short)0, acc0, false, false);
            acc1 = __builtin_amdgcn_wmma_f32_16x16x32_bf16(
                       false, afrag[1][ks], false, bf, (short)0, acc1, false, false);
        }

        // ---- epilogue: h_new = tanh(acc + W_hx*x_t + b_h), write to next buffer ----
        // C/D layout: VGPR j holds (M = tilebase + 8*sel + j, N = bcol) -> 8
        // consecutive h rows per lane => single 16B ds_store per tile.
        #pragma unroll
        for (int tt = 0; tt < MT_PER_WAVE; ++tt) {
            const v8f a = (tt == 0) ? acc0 : acc1;
            v8bf hnew;
            #pragma unroll
            for (int j = 0; j < 8; ++j) {
                const float w  = (j < 4) ? whxA[tt][j] : whxB[tt][j - 4];
                const float bb = (j < 4) ? bhA[tt][j]  : bhB[tt][j - 4];
                hnew[j] = (__bf16)fast_tanh(a[j] + w * xt + bb);
            }
            const int mb = (wave * MT_PER_WAVE + tt) * 16 + sel * 8;
            *(v8bf*)&hbuf[nxt][bcol][mb] = hnew;  // ds_store_b128
        }
        xt = xnext;
        __syncthreads();  // publish h_new before anyone reads it next step
    }

    // ---- projection: p[b,c] = sum_h h_T[h,b] * W_ph[c,h] + bp[c] ----
    // Final h is in hbuf[0] (T even). 16 cols x 10 classes = 160 outputs/WG.
    if (tid < NB * CLS) {
        const int b = tid / CLS;
        const int c = tid % CLS;
        float s = bp[c];
        const __bf16* hv = hbuf[0][b];
        const float*  wr = &Wph[c * HID];
        #pragma unroll 8
        for (int k = 0; k < HID; ++k)
            s += (float)hv[k] * wr[k];
        out[(bbase + b) * CLS + c] = s;
    }
}

extern "C" void kernel_launch(void* const* d_in, const int* in_sizes, int n_in,
                              void* d_out, int out_size, void* d_ws, size_t ws_size,
                              hipStream_t stream) {
    const float* x   = (const float*)d_in[0];  // [512, 2048]
    const float* Whx = (const float*)d_in[1];  // [256]
    const float* Whh = (const float*)d_in[2];  // [256, 256]
    const float* Wph = (const float*)d_in[3];  // [10, 256]
    const float* bh  = (const float*)d_in[4];  // [256]
    const float* bp  = (const float*)d_in[5];  // [10]
    float* out = (float*)d_out;                // [512, 10]

    VanillaRNN_cdna5_kernel<<<BATCH / NB, 256, 0, stream>>>(
        x, Whx, Whh, Wph, bh, bp, out);
}
